// SequenceLayer_4071628997167
// MI455X (gfx1250) — matrix operand order repaired and checked by reading, not compiled, not run on hardware
//
#include <hip/hip_runtime.h>
#include <hip/hip_bf16.h>
#include <math.h>

// Problem constants (from reference)
#define BB   16
#define LL   8192
#define HH   128
#define PP   128
#define LC   32            // scan chunk length
#define NCH  (LL / LC)     // 256 chunks

typedef __attribute__((ext_vector_type(16))) _Float16 v16h;
typedef __attribute__((ext_vector_type(8)))  float    v8f;

union Frag16 { v16h v; uint4 q[2]; };

// Workspace layout (bytes)
#define OFF_W1RE  ((size_t)0)                    // 128*128 f16 = 32KB
#define OFF_W1IM  ((size_t)(32 * 1024))          // 32KB
#define OFF_W2    ((size_t)(64 * 1024))          // 128*256 f16 = 64KB
#define OFF_APOW  ((size_t)(192 * 1024))         // 32*128 float2 = 32KB
#define OFF_ABAR  ((size_t)(240 * 1024))         // 128 float2 = 1KB
#define OFF_EL    ((size_t)(1  << 20))           // B*NCH*P float2 = 4MB
#define OFF_PREF  ((size_t)(5  << 20))           // 4MB
#define OFF_XS    ((size_t)(9  << 20))           // B*L*256 f16 = 64MB

__device__ __forceinline__ float2 cmul(float2 a, float2 b) {
    return make_float2(a.x * b.x - a.y * b.y, a.x * b.y + a.y * b.x);
}

// ---------------------------------------------------------------------------
// k0: discretization + weight packing (1 block, 128 threads)
// ---------------------------------------------------------------------------
__global__ void k0_precompute(const float* __restrict__ Lre, const float* __restrict__ Lim,
                              const float* __restrict__ Bre, const float* __restrict__ Bim,
                              const float* __restrict__ Cre, const float* __restrict__ Cim,
                              const float* __restrict__ logstep, char* __restrict__ ws) {
    const int p = threadIdx.x;                    // 0..127
    _Float16* W1re = (_Float16*)(ws + OFF_W1RE);
    _Float16* W1im = (_Float16*)(ws + OFF_W1IM);
    _Float16* W2   = (_Float16*)(ws + OFF_W2);
    float2*   Apow = (float2*)(ws + OFF_APOW);
    float2*   Abar = (float2*)(ws + OFF_ABAR);

    const float lr = Lre[p], li = Lim[p];
    const float st = __expf(logstep[p]);
    const float mag = __expf(lr * st);
    const float2 A = make_float2(mag * __cosf(li * st), mag * __sinf(li * st));
    // coef = (A - 1) / Lambda
    const float den = lr * lr + li * li;
    const float nre = A.x - 1.0f, nim = A.y;
    const float2 coef = make_float2((nre * lr + nim * li) / den,
                                    (nim * lr - nre * li) / den);
    for (int h = 0; h < HH; ++h) {
        const float br = Bre[p * HH + h], bi = Bim[p * HH + h];
        W1re[p * HH + h] = (_Float16)(coef.x * br - coef.y * bi);
        W1im[p * HH + h] = (_Float16)(coef.x * bi + coef.y * br);
    }
    Abar[p] = A;
    float2 ap = A;
    for (int t = 0; t < LC; ++t) {               // Apow[t] = A^(t+1)
        Apow[t * PP + p] = ap;
        ap = cmul(ap, A);
    }
    // W2[h][k]: k<128 -> 2*C_re[h][k], k>=128 -> -2*C_im[h][k-128]
    const int h = p;
    for (int pp = 0; pp < PP; ++pp) {
        W2[h * 256 + pp]       = (_Float16)( 2.0f * Cre[h * PP + pp]);
        W2[h * 256 + 128 + pp] = (_Float16)(-2.0f * Cim[h * PP + pp]);
    }
}

// ---------------------------------------------------------------------------
// k1: Bu GEMM (f16 WMMA, f32 acc) + chunk-local scan (f32)
// grid = (NCH, B), 256 threads (8 waves)
// ---------------------------------------------------------------------------
__global__ __launch_bounds__(256) void k1_bu_scan(const float* __restrict__ u,
                                                  char* __restrict__ ws) {
    const int b = blockIdx.y, c = blockIdx.x;
    const int tid = threadIdx.x;
    __shared__ _Float16 sh_u[LC][HH];            // 8KB
    __shared__ float2   sh_bu[LC][PP];           // 32KB

    // 1) stage u chunk -> f16 LDS
    const size_t ubase = ((size_t)b * LL + (size_t)c * LC) * HH;
    const float4* u4 = (const float4*)(u + ubase);
    for (int i = tid; i < LC * HH / 4; i += 256) {
        const float4 v = u4[i];
        const int f = i * 4;
        _Float16* d = &sh_u[f >> 7][f & 127];
        d[0] = (_Float16)v.x; d[1] = (_Float16)v.y;
        d[2] = (_Float16)v.z; d[3] = (_Float16)v.w;
    }
    __syncthreads();

    // 2) Bu tiles: wave w owns p-columns [16w,16w+16)
    const _Float16* W1re = (const _Float16*)(ws + OFF_W1RE);
    const _Float16* W1im = (const _Float16*)(ws + OFF_W1IM);
    const int w = tid >> 5, ln = tid & 31;
    const int col = ln & 15, ksel = ln >> 4;
    const int pbase = w * 16;
#pragma unroll
    for (int mt = 0; mt < 2; ++mt) {
        v8f ar = {}; v8f ai = {};
#pragma unroll
        for (int kb = 0; kb < HH; kb += 32) {
            Frag16 a, bre, bim;
            const int row = mt * 16 + col;       // A: lane -> M row
            a.q[0] = *(const uint4*)&sh_u[row][kb + ksel * 8];
            a.q[1] = *(const uint4*)&sh_u[row][kb + 16 + ksel * 8];
            const _Float16* wr = W1re + (size_t)(pbase + col) * HH + kb + ksel * 8;
            bre.q[0] = *(const uint4*)wr;       bre.q[1] = *(const uint4*)(wr + 16);
            const _Float16* wi = W1im + (size_t)(pbase + col) * HH + kb + ksel * 8;
            bim.q[0] = *(const uint4*)wi;       bim.q[1] = *(const uint4*)(wi + 16);
            ar = __builtin_amdgcn_wmma_f32_16x16x32_f16(false, a.v, false, bre.v,
                                                        (short)0, ar, false, false);
            ai = __builtin_amdgcn_wmma_f32_16x16x32_f16(false, a.v, false, bim.v,
                                                        (short)0, ai, false, false);
        }
#pragma unroll
        for (int r = 0; r < 8; ++r) {            // D layout: M = r + 8*ksel
            const int t = mt * 16 + r + 8 * ksel;
            sh_bu[t][pbase + col] = make_float2(ar[r], ai[r]);
        }
    }
    __syncthreads();

    // 3) chunk-local scan (f32), write x_local as f16 [re|im] rows, end state f32
    if (tid < PP) {
        const int p = tid;
        const float2 A = ((const float2*)(ws + OFF_ABAR))[p];
        float2 s = make_float2(0.f, 0.f);
        _Float16* xs = (_Float16*)(ws + OFF_XS);
        const size_t mrow = (size_t)b * LL + (size_t)c * LC;
        for (int t = 0; t < LC; ++t) {
            const float2 bu = sh_bu[t][p];
            s = make_float2(A.x * s.x - A.y * s.y + bu.x,
                            A.x * s.y + A.y * s.x + bu.y);
            _Float16* row = xs + (mrow + t) * 256;
            row[p]       = (_Float16)s.x;
            row[128 + p] = (_Float16)s.y;
        }
        ((float2*)(ws + OFF_EL))[((size_t)b * NCH + c) * PP + p] = s;
    }
}

// ---------------------------------------------------------------------------
// k2: inter-chunk carry scan, (b,p) parallel, sequential over 256 chunks
// ---------------------------------------------------------------------------
__global__ void k2_carry(char* __restrict__ ws) {
    const int g = blockIdx.x * blockDim.x + threadIdx.x;  // 0..2047
    const int b = g >> 7, p = g & 127;
    const float2 Alc = ((const float2*)(ws + OFF_APOW))[(LC - 1) * PP + p]; // A^LC
    const float2* el = (const float2*)(ws + OFF_EL);
    float2* pref = (float2*)(ws + OFF_PREF);
    float2 s = make_float2(0.f, 0.f);
    for (int c = 0; c < NCH; ++c) {
        const size_t idx = ((size_t)b * NCH + c) * PP + p;
        pref[idx] = s;
        const float2 e = el[idx];
        s = make_float2(Alc.x * s.x - Alc.y * s.y + e.x,
                        Alc.x * s.y + Alc.y * s.x + e.y);
    }
}

// ---------------------------------------------------------------------------
// k3: apply carries + readout GEMM (K=256 f16 WMMA) + D*u epilogue
// grid = (L/16, B), 256 threads (8 waves; wave w -> h tile [16w,16w+16))
// ---------------------------------------------------------------------------
__global__ __launch_bounds__(256) void k3_readout(const float* __restrict__ u,
                                                  const float* __restrict__ Dv,
                                                  float* __restrict__ y,
                                                  char* __restrict__ ws) {
    const int b = blockIdx.y;
    const int lb = blockIdx.x * 16;
    const int tid = threadIdx.x;
    __shared__ _Float16 sh_x[16][256];           // 8KB

    const int cIdx = lb / LC, tlb = lb % LC;
    // 1) stage corrected x: thread owns column k for all 16 rows
    {
        const int k = tid, p = k & 127, isIm = k >> 7;
        const float2 s = ((const float2*)(ws + OFF_PREF))[((size_t)b * NCH + cIdx) * PP + p];
        const float2* Apow = (const float2*)(ws + OFF_APOW);
        const _Float16* xs = (const _Float16*)(ws + OFF_XS);
        const size_t mbase = (size_t)b * LL + lb;
        for (int r = 0; r < 16; ++r) {
            const float2 ap = Apow[(tlb + r) * PP + p];      // A^(t_local+1)
            const float2 corr = cmul(ap, s);
            const float v = (float)xs[(mbase + r) * 256 + k] + (isIm ? corr.y : corr.x);
            sh_x[r][k] = (_Float16)v;
        }
    }
    __syncthreads();

    // 2) y tile = x(16x256) * W2^T(256x16) via 8 WMMAs
    const _Float16* W2 = (const _Float16*)(ws + OFF_W2);
    const int w = tid >> 5, ln = tid & 31;
    const int col = ln & 15, ksel = ln >> 4;
    const int hbase = w * 16;
    v8f acc = {};
#pragma unroll
    for (int kb = 0; kb < 256; kb += 32) {
        Frag16 a, bw;
        a.q[0] = *(const uint4*)&sh_x[col][kb + ksel * 8];
        a.q[1] = *(const uint4*)&sh_x[col][kb + 16 + ksel * 8];
        const _Float16* wr = W2 + (size_t)(hbase + col) * 256 + kb + ksel * 8;
        bw.q[0] = *(const uint4*)wr;  bw.q[1] = *(const uint4*)(wr + 16);
        acc = __builtin_amdgcn_wmma_f32_16x16x32_f16(false, a.v, false, bw.v,
                                                     (short)0, acc, false, false);
    }

    // 3) epilogue: + D*u
    const int h = hbase + col;
    const float Dh = Dv[h];
    const size_t mbase = (size_t)b * LL + lb;
#pragma unroll
    for (int r = 0; r < 8; ++r) {
        const int m = r + 8 * ksel;
        const size_t off = (mbase + m) * HH + h;
        y[off] = acc[r] + Dh * u[off];
    }
}

// ---------------------------------------------------------------------------
extern "C" void kernel_launch(void* const* d_in, const int* in_sizes, int n_in,
                              void* d_out, int out_size, void* d_ws, size_t ws_size,
                              hipStream_t stream) {
    const float* u       = (const float*)d_in[0];
    const float* Lre     = (const float*)d_in[1];
    const float* Lim     = (const float*)d_in[2];
    const float* Bre     = (const float*)d_in[3];
    const float* Bim     = (const float*)d_in[4];
    const float* Cre     = (const float*)d_in[5];
    const float* Cim     = (const float*)d_in[6];
    const float* Dv      = (const float*)d_in[7];
    const float* logstep = (const float*)d_in[8];
    float* y = (float*)d_out;
    char*  ws = (char*)d_ws;

    k0_precompute<<<1, 128, 0, stream>>>(Lre, Lim, Bre, Bim, Cre, Cim, logstep, ws);
    k1_bu_scan<<<dim3(NCH, BB), 256, 0, stream>>>(u, ws);
    k2_carry<<<(BB * PP) / 256, 256, 0, stream>>>(ws);
    k3_readout<<<dim3(LL / 16, BB), 256, 0, stream>>>(u, Dv, y, ws);
}
